// WD_GCN_7327214207510
// MI455X (gfx1250) — matrix operand order repaired and checked by reading, not compiled, not run on hardware
//
#include <hip/hip_runtime.h>
#include <hip/hip_bf16.h>
#include <math.h>

// ---------------- problem constants ----------------
#define TT   8
#define NN   10000
#define FIN  128
#define HH   64
#define EE   160000
#define TN   (TT * NN)        // 80000 rows
#define G4H  (4 * HH)         // 256 gates

typedef __attribute__((ext_vector_type(16))) __bf16 v16bf;
typedef __attribute__((ext_vector_type(8)))  float  v8f;

// split a float into bf16 hi/lo parts (hi + lo ~ 16 mantissa bits)
__device__ __forceinline__ void bsplit(float v, __bf16& hi, __bf16& lo) {
    hi = (__bf16)v;
    lo = (__bf16)(v - (float)hi);
}

// ---- gfx1250 async global->LDS copy (ASYNCcnt path, ISA 08 §4) ----
// VDST = VGPR holding LDS byte address (ISA: LDS addr = flat addr[31:0]),
// VADDR = 64-bit global address, saddr = off.
__device__ __forceinline__ void async_b128(const void* g, void* l) {
    unsigned lds_addr = (unsigned)(unsigned long long)l;
    asm volatile("global_load_async_to_lds_b128 %0, %1, off"
                 :
                 : "v"(lds_addr), "v"(g)
                 : "memory");
}
__device__ __forceinline__ void async_wait0() {
    asm volatile("s_wait_asynccnt 0x0" ::: "memory");
}

// =====================================================================
// Kernel 1: C1[80000,64] = x[80000,128] @ W_gcn[128,64]
// grid = 5000 (16 rows each), block = 128 (4 waves, one 16-col tile each)
// =====================================================================
__global__ void gcn_gemm_kernel(const float* __restrict__ x,
                                const float* __restrict__ Wg,
                                float* __restrict__ c1) {
    __shared__ float Bs[FIN * HH];   // 32 KB: whole W_gcn
    __shared__ float As[16 * FIN];   //  8 KB: 16 rows of x

    const int tid = threadIdx.x;
    const int m0  = blockIdx.x * 16;

    // async-copy W_gcn (32KB) and the A tile (8KB) into LDS, 16B per op
    {
        const char* gw = (const char*)Wg;
        char*       lw = (char*)Bs;
        for (int off = tid * 16; off < FIN * HH * 4; off += 128 * 16)
            async_b128(gw + off, lw + off);
        const char* ga = (const char*)(x + (size_t)m0 * FIN);
        char*       la = (char*)As;
        for (int off = tid * 16; off < 16 * FIN * 4; off += 128 * 16)
            async_b128(ga + off, la + off);
        async_wait0();
    }
    __syncthreads();

    const int lane = tid & 31;
    const int wave = tid >> 5;
    const int n0   = wave * 16;          // 4 waves cover the 64 cols
    const int row  = lane & 15;          // A row within tile
    const int col  = lane & 15;          // B/D column within tile
    const int hi16 = lane >> 4;          // 0 or 1
    const int akb  = hi16 << 3;          // unused; A also uses bkb (0 / 16)
    const int bkb  = hi16 << 4;          // B K-half offset (0 / 16)

    v8f acc = {};
    for (int k0 = 0; k0 < FIN; k0 += 32) {
        v16bf ah, al, bh, bl;
#pragma unroll
        for (int e = 0; e < 16; ++e) {
            const int ka  = k0 + bkb + e;            // A layout
            const int kbK = k0 + bkb + e;            // B layout
            __bf16 h, l;
            bsplit(As[row * FIN + ka], h, l); ah[e] = h; al[e] = l;
            bsplit(Bs[kbK * HH + n0 + col], h, l); bh[e] = h; bl[e] = l;
        }
        acc = __builtin_amdgcn_wmma_f32_16x16x32_bf16(false, al, false, bh, (short)0, acc, false, false);
        acc = __builtin_amdgcn_wmma_f32_16x16x32_bf16(false, ah, false, bl, (short)0, acc, false, false);
        acc = __builtin_amdgcn_wmma_f32_16x16x32_bf16(false, ah, false, bh, (short)0, acc, false, false);
    }

    const int mbase = m0 + (hi16 << 3);
#pragma unroll
    for (int r = 0; r < 8; ++r)
        c1[(size_t)(mbase + r) * HH + n0 + col] = acc[r];
}

// =====================================================================
// Degree / normalization kernels
// =====================================================================
__global__ void deg_init_kernel(float* __restrict__ deg) {
    const int i = blockIdx.x * 256 + threadIdx.x;
    if (i < TN) deg[i] = 1.0f;                    // self loop
}

__global__ void deg_count_kernel(const long long* __restrict__ ei,
                                 float* __restrict__ deg) {
    const size_t i = (size_t)blockIdx.x * 256 + threadIdx.x;
    if (i >= (size_t)TT * EE) return;
    const int t = (int)(i / EE);
    const int e = (int)(i % EE);
    const int dst = (int)ei[(size_t)t * 2 * EE + EE + e];
    atomicAdd(&deg[t * NN + dst], 1.0f);
}

__global__ void rsqrt_kernel(float* __restrict__ deg) {
    const int i = blockIdx.x * 256 + threadIdx.x;
    if (i < TN) deg[i] = rsqrtf(deg[i]);
}

// self-loop contribution: g = c1 * dinv^2
__global__ void self_init_kernel(const float* __restrict__ c1,
                                 const float* __restrict__ dinv,
                                 float* __restrict__ g) {
    const size_t i = (size_t)blockIdx.x * 256 + threadIdx.x;
    if (i >= (size_t)TN * HH) return;
    const size_t node = i >> 6;
    const float d = dinv[node];
    g[i] = c1[i] * d * d;
}

// edge scatter: g[t][dst] += c1[t][src] * dinv[src]*dinv[dst]
// 16 threads per edge, 4 features each
__global__ void edge_scatter_kernel(const long long* __restrict__ ei,
                                    const float* __restrict__ c1,
                                    const float* __restrict__ dinv,
                                    float* __restrict__ g) {
    const size_t i = (size_t)blockIdx.x * 256 + threadIdx.x;
    if (i >= (size_t)TT * EE * 16) return;
    const int sub     = (int)(i & 15);
    const size_t eidx = i >> 4;
    const int t = (int)(eidx / EE);
    const int e = (int)(eidx % EE);
    const long long* base = ei + (size_t)t * 2 * EE;
    const int src = (int)base[e];
    const int dst = (int)base[EE + e];
    const float norm = dinv[t * NN + src] * dinv[t * NN + dst];
    const float4 v = *(const float4*)(c1 + ((size_t)t * NN + src) * HH + sub * 4);
    float* dp = g + ((size_t)t * NN + dst) * HH + sub * 4;
    atomicAdd(dp + 0, v.x * norm);
    atomicAdd(dp + 1, v.y * norm);
    atomicAdd(dp + 2, v.z * norm);
    atomicAdd(dp + 3, v.w * norm);
}

// =====================================================================
// Kernel: Xih[80000,256] = relu(g + b_gcn) @ W_ih^T + (b_ih + b_hh)
// grid = 5000, block = 512 (16 waves, one 16-col tile each)
// =====================================================================
__global__ void ih_gemm_kernel(const float* __restrict__ g,
                               const float* __restrict__ bg,
                               const float* __restrict__ Wih,
                               const float* __restrict__ bih,
                               const float* __restrict__ bhh,
                               float* __restrict__ xih) {
    __shared__ float As[16 * HH];    // 4 KB: relu(g + b_gcn) tile

    const int tid = threadIdx.x;
    const int m0  = blockIdx.x * 16;

    for (int i = tid; i < 16 * HH; i += 512) {
        float v = g[(size_t)m0 * HH + i] + bg[i & (HH - 1)];
        As[i] = v > 0.f ? v : 0.f;
    }
    __syncthreads();

    const int lane = tid & 31;
    const int wave = tid >> 5;
    const int n0   = wave * 16;          // 16 waves cover 256 cols
    const int row  = lane & 15;
    const int col  = n0 + (lane & 15);
    const int hi16 = lane >> 4;
    const int akb  = hi16 << 3;
    const int bkb  = hi16 << 4;

    v8f acc = {};
    for (int k0 = 0; k0 < HH; k0 += 32) {
        v16bf ah, al, bh, bl;
#pragma unroll
        for (int e = 0; e < 16; ++e) {
            const int ka  = k0 + bkb + e;
            const int kbK = k0 + bkb + e;
            __bf16 h, l;
            bsplit(As[row * HH + ka], h, l); ah[e] = h; al[e] = l;
            // B[k][col] = W_ih[col][k]  (W_ih is [256,64] row-major)
            bsplit(Wih[(size_t)col * HH + kbK], h, l); bh[e] = h; bl[e] = l;
        }
        acc = __builtin_amdgcn_wmma_f32_16x16x32_bf16(false, al, false, bh, (short)0, acc, false, false);
        acc = __builtin_amdgcn_wmma_f32_16x16x32_bf16(false, ah, false, bl, (short)0, acc, false, false);
        acc = __builtin_amdgcn_wmma_f32_16x16x32_bf16(false, ah, false, bh, (short)0, acc, false, false);
    }

    const float bb = bih[col] + bhh[col];
    const int mbase = m0 + (hi16 << 3);
#pragma unroll
    for (int r = 0; r < 8; ++r)
        xih[(size_t)(mbase + r) * G4H + col] = acc[r] + bb;
}

// =====================================================================
// Serial LSTM scan over N=10000 steps; 8 independent graphs = 8 blocks.
// block = 256: thread j owns gate j (row j of W_hh in registers).
// Prefetch next step's gate-input row to hide its latency under the
// 64-FMA recurrence chain (latency-bound kernel).
// =====================================================================
__global__ void lstm_scan_kernel(const float* __restrict__ xih,
                                 const float* __restrict__ Whh,
                                 float* __restrict__ out) {
    __shared__ float hl[HH];
    __shared__ float sg[G4H];

    const int t = blockIdx.x;
    const int j = threadIdx.x;       // 0..255

    float w[HH];
#pragma unroll
    for (int k = 0; k < HH; ++k) w[k] = Whh[j * HH + k];

    if (j < HH) hl[j] = 0.f;
    float c = 0.f;
    __syncthreads();

    const float* xp = xih + (size_t)t * NN * G4H;
    float* op = out + (size_t)t * NN * HH;

    for (int n = 0; n < NN; ++n) {
        const int np = (n + 1 < NN) ? (n + 1) : n;
        __builtin_prefetch(&xp[(size_t)np * G4H + j], 0, 0);   // global_prefetch_b8

        float acc = xp[(size_t)n * G4H + j];
#pragma unroll
        for (int k = 0; k < HH; ++k) acc += hl[k] * w[k];
        sg[j] = acc;
        __syncthreads();
        if (j < HH) {
            const float iv = 1.f / (1.f + __expf(-sg[j]));
            const float fv = 1.f / (1.f + __expf(-sg[HH + j]));
            const float gv = tanhf(sg[2 * HH + j]);
            const float ov = 1.f / (1.f + __expf(-sg[3 * HH + j]));
            c = fv * c + iv * gv;
            const float hn = ov * tanhf(c);
            hl[j] = hn;
            op[(size_t)n * HH + j] = hn;
        }
        __syncthreads();
    }
}

// =====================================================================
extern "C" void kernel_launch(void* const* d_in, const int* in_sizes, int n_in,
                              void* d_out, int out_size, void* d_ws, size_t ws_size,
                              hipStream_t stream) {
    const float*     x    = (const float*)d_in[0];
    const long long* ei   = (const long long*)d_in[1];   // int64 edge_index
    const float*     Wg   = (const float*)d_in[2];
    const float*     bg   = (const float*)d_in[3];
    const float*     Wih  = (const float*)d_in[4];
    const float*     Whh  = (const float*)d_in[5];
    const float*     bih  = (const float*)d_in[6];
    const float*     bhh  = (const float*)d_in[7];
    float*           out  = (float*)d_out;

    // workspace carve-up (floats)
    float* ws   = (float*)d_ws;
    float* c1   = ws;                          // [80000 x 64]
    float* g    = c1 + (size_t)TN * HH;        // [80000 x 64]
    float* dinv = g + (size_t)TN * HH;         // [80000]  (deg, then rsqrt in-place)
    float* xih  = dinv + TN;                   // [80000 x 256]

    // 1) GCN linear transform (WMMA, async LDS staging)
    gcn_gemm_kernel<<<TN / 16, 128, 0, stream>>>(x, Wg, c1);

    // 2) degrees with self loops -> dinv
    deg_init_kernel<<<(TN + 255) / 256, 256, 0, stream>>>(dinv);
    deg_count_kernel<<<(TT * EE + 255) / 256, 256, 0, stream>>>(ei, dinv);
    rsqrt_kernel<<<(TN + 255) / 256, 256, 0, stream>>>(dinv);

    // 3) normalized propagation (self loop init + edge scatter)
    self_init_kernel<<<((size_t)TN * HH + 255) / 256, 256, 0, stream>>>(c1, dinv, g);
    edge_scatter_kernel<<<((size_t)TT * EE * 16 + 255) / 256, 256, 0, stream>>>(ei, c1, dinv, g);

    // 4) bias + ReLU folded into LSTM input GEMM (WMMA), biases folded in epilogue
    ih_gemm_kernel<<<TN / 16, 512, 0, stream>>>(g, bg, Wih, bih, bhh, xih);

    // 5) serial recurrence (8 independent graphs)
    lstm_scan_kernel<<<TT, 256, 0, stream>>>(xih, Whh, out);
}
